// MultiHeadLatentAttention_81157702025387
// MI455X (gfx1250) — matrix operand order repaired and checked by reading, not compiled, not run on hardware
//
#include <hip/hip_runtime.h>
#include <hip/hip_bf16.h>

// ---------------------------------------------------------------------------
// MLA forward for MI455X (gfx1250): bf16 WMMA everywhere, flash attention,
// async global->LDS tile streaming (ASYNCcnt) with double buffering.
// ---------------------------------------------------------------------------

#define BB    2
#define SS    2048
#define DIN   2048
#define DOUT  2048
#define LAT   512
#define HEADS 16
#define HD    128

typedef __attribute__((ext_vector_type(16))) __bf16     bf16x16;
typedef __attribute__((ext_vector_type(8)))  float      f32x8;
typedef __attribute__((ext_vector_type(4)))  float      f32x4;
typedef __attribute__((ext_vector_type(4)))  unsigned   u32x4;
typedef int v4i_vs __attribute__((vector_size(16)));  // builtin's pointee type

union FragU { bf16x16 v; u32x4 q[2]; };

#define AS1 __attribute__((address_space(1)))
#define AS3 __attribute__((address_space(3)))

#if defined(__HIP_DEVICE_COMPILE__) &&                                   \
    __has_builtin(__builtin_amdgcn_global_load_async_to_lds_b128) &&     \
    __has_builtin(__builtin_amdgcn_s_wait_asynccnt)
#define USE_ASYNC_LDS 1
#else
#define USE_ASYNC_LDS 0
#endif

// Stage 16 bytes global -> LDS. Async (no VGPR data path) when available.
__device__ __forceinline__ void stage16(const __bf16* g, __bf16* l) {
#if USE_ASYNC_LDS
  __builtin_amdgcn_global_load_async_to_lds_b128((AS1 v4i_vs*)g,
                                                 (AS3 v4i_vs*)l, 0, 0);
#else
  *(u32x4*)l = *(const u32x4*)g;
#endif
}
__device__ __forceinline__ void stage_wait() {
#if USE_ASYNC_LDS
  __builtin_amdgcn_s_wait_asynccnt(0);
#endif
}

__device__ __forceinline__ unsigned short f2bf_bits(float f) {
  unsigned u = __float_as_uint(f);
  unsigned r = (u + 0x7FFFu + ((u >> 16) & 1u)) >> 16;
  return (unsigned short)r;
}
__device__ __forceinline__ __bf16 f2bf(float f) {
  unsigned short s = f2bf_bits(f);
  return __builtin_bit_cast(__bf16, s);
}
__device__ __forceinline__ float bf2f(__bf16 b) {
  unsigned short s = __builtin_bit_cast(unsigned short, b);
  return __uint_as_float(((unsigned)s) << 16);
}

// A-fragment (16x32 bf16): lane l<16 row=l holds K 0..7 (q0) and 16..23 (q1);
// lane l+16 holds K 8..15 and 24..31.   (ISA 7.12.2 16-bit A layout)
__device__ __forceinline__ bf16x16 fragA(const __bf16* rowp, int hi) {
  FragU f;
  f.q[0] = *(const u32x4*)(rowp + hi * 8);
  f.q[1] = *(const u32x4*)(rowp + 16 + hi * 8);
  return f.v;
}
// B-fragment (32x16 bf16) from Bt stored [N][K]: lane l<16 col=l holds K 0..7
// and 16..23; lane l+16 holds K 8..15 and 24..31 (same K order as fragA).
__device__ __forceinline__ bf16x16 fragB(const __bf16* rowp, int hi) {
  FragU f;
  f.q[0] = *(const u32x4*)(rowp + hi * 8);
  f.q[1] = *(const u32x4*)(rowp + 16 + hi * 8);
  return f.v;
}

__device__ __forceinline__ f32x8 wmma_bf16(bf16x16 a, bf16x16 b, f32x8 c) {
  return __builtin_amdgcn_wmma_f32_16x16x32_bf16(false, a, false, b,
                                                 (short)0, c, false, false);
}

// ---------------------------------------------------------------------------
// fp32 -> bf16 elementwise (4 elems/thread)
// ---------------------------------------------------------------------------
__global__ void cvt_bf16_kernel(const float* __restrict__ in,
                                __bf16* __restrict__ out, int n4) {
  int i = blockIdx.x * 256 + threadIdx.x;
  if (i >= n4) return;
  f32x4 f = ((const f32x4*)in)[i];
  unsigned long long p = (unsigned long long)f2bf_bits(f.x)
                       | ((unsigned long long)f2bf_bits(f.y) << 16)
                       | ((unsigned long long)f2bf_bits(f.z) << 32)
                       | ((unsigned long long)f2bf_bits(f.w) << 48);
  ((unsigned long long*)out)[i] = p;
}

// ---------------------------------------------------------------------------
// fp32 W[K][N] -> bf16 Wt[N][K]  (tiled transpose)
// ---------------------------------------------------------------------------
__global__ void transpose_bf16_kernel(const float* __restrict__ W,
                                      __bf16* __restrict__ Wt, int K, int N) {
  __shared__ float tile[32][33];
  int nb = blockIdx.x * 32, kb = blockIdx.y * 32;
  int tx = threadIdx.x & 31, ty = threadIdx.x >> 5;
#pragma unroll
  for (int i = ty; i < 32; i += 8)
    tile[i][tx] = W[(size_t)(kb + i) * N + nb + tx];
  __syncthreads();
#pragma unroll
  for (int i = ty; i < 32; i += 8)
    Wt[(size_t)(nb + i) * K + kb + tx] = f2bf(tile[tx][i]);
}

// ---------------------------------------------------------------------------
// GEMM: C[M][N] = A[M][K] * Bt[N][K]^T   (bf16 in, f32 accumulate)
// block tile 128x128, 8 waves each 32x64, K-tile 64 (2 WMMA K-steps),
// double-buffered LDS fed by async global->LDS copies.
// ---------------------------------------------------------------------------
#define LDT 80  // padded LDS row stride for 64-wide K tile (160B, 16B-aligned)

template <int OUT_BF16>
__global__ __launch_bounds__(256)
void gemm_bf16_kernel(const __bf16* __restrict__ A,
                      const __bf16* __restrict__ Bt,
                      void* __restrict__ C, int M, int N, int K) {
  __shared__ __bf16 sA[2][128 * LDT];
  __shared__ __bf16 sB[2][128 * LDT];

  const int tid  = threadIdx.x;
  const int w    = tid >> 5;
  const int lane = tid & 31;
  const int l    = lane & 15;
  const int hi   = lane >> 4;
  const int wm   = (w >> 1) * 32;
  const int wn   = (w & 1) * 64;
  const int mBase = blockIdx.y * 128;
  const int nBase = blockIdx.x * 128;

  f32x8 zero = {0.f, 0.f, 0.f, 0.f, 0.f, 0.f, 0.f, 0.f};
  f32x8 acc[2][4];
#pragma unroll
  for (int mi = 0; mi < 2; ++mi)
#pragma unroll
    for (int ni = 0; ni < 4; ++ni) acc[mi][ni] = zero;

  // 128 rows x 64 cols = 1024 16B-chunks per matrix; 4 chunks per thread.
  auto stageTile = [&](int k0, int bf) {
#pragma unroll
    for (int i = 0; i < 4; ++i) {
      int c = tid + i * 256, row = c >> 3, cc = (c & 7) * 8;
      stage16(A + (size_t)(mBase + row) * K + k0 + cc, &sA[bf][row * LDT + cc]);
      stage16(Bt + (size_t)(nBase + row) * K + k0 + cc, &sB[bf][row * LDT + cc]);
    }
  };

  stageTile(0, 0);
  stage_wait();
  __syncthreads();

  int buf = 0;
  for (int k0 = 0; k0 < K; k0 += 64) {
    if (k0 + 64 < K) stageTile(k0 + 64, buf ^ 1);  // streams in under WMMAs
#pragma unroll
    for (int kk = 0; kk < 64; kk += 32) {
      bf16x16 af[2], bfg[4];
#pragma unroll
      for (int mi = 0; mi < 2; ++mi)
        af[mi] = fragA(&sA[buf][(wm + mi * 16 + l) * LDT + kk], hi);
#pragma unroll
      for (int ni = 0; ni < 4; ++ni)
        bfg[ni] = fragB(&sB[buf][(wn + ni * 16 + l) * LDT + kk], hi);
#pragma unroll
      for (int mi = 0; mi < 2; ++mi)
#pragma unroll
        for (int ni = 0; ni < 4; ++ni)
          acc[mi][ni] = wmma_bf16(af[mi], bfg[ni], acc[mi][ni]);
    }
    stage_wait();
    __syncthreads();
    buf ^= 1;
  }

  // C-fragment: row = hi*8 + vgpr, col = l  (ISA 7.12.2 C/D layout)
#pragma unroll
  for (int mi = 0; mi < 2; ++mi)
#pragma unroll
    for (int ni = 0; ni < 4; ++ni)
#pragma unroll
      for (int r = 0; r < 8; ++r) {
        int row = mBase + wm + mi * 16 + hi * 8 + r;
        int col = nBase + wn + ni * 16 + l;
        float v = acc[mi][ni][r];
        if (OUT_BF16) ((__bf16*)C)[(size_t)row * N + col] = f2bf(v);
        else          ((float*)C)[(size_t)row * N + col]  = v;
      }
}

// ---------------------------------------------------------------------------
// RoPE on Q: qp bf16 [B][S][DOUT] -> qh bf16 [B][H][S][HD]
// ---------------------------------------------------------------------------
__global__ void rope_q_kernel(const __bf16* __restrict__ qp,
                              const float* __restrict__ pe,
                              __bf16* __restrict__ qh) {
  int tid = blockIdx.x * 256 + threadIdx.x;
  int d2 = tid & 63;
  int h  = (tid >> 6) & 15;
  int s  = (tid >> 10) & 2047;
  int b  = (tid >> 21) & 1;
  size_t in_off = ((size_t)(b * SS + s)) * DOUT + h * HD + d2;
  float x1 = bf2f(qp[in_off]);
  float x2 = bf2f(qp[in_off + 64]);
  float p1 = pe[s * HD + d2], p2 = pe[s * HD + d2 + 64];
  float o1 = x1 * __cosf(p1) - x2 * __sinf(p1);
  float o2 = x2 * __cosf(p2) + x1 * __sinf(p2);
  size_t out_off = (((size_t)(b * HEADS + h)) * SS + s) * HD + d2;
  qh[out_off]      = f2bf(o1);
  qh[out_off + 64] = f2bf(o2);
}

// ---------------------------------------------------------------------------
// Split kv -> RoPE(K) head-major + V transposed:
// kvp bf16 [B][S][2*DOUT] -> kh [B][H][S][HD], vt [B][H][HD][S]
// ---------------------------------------------------------------------------
__global__ void rope_kv_kernel(const __bf16* __restrict__ kvp,
                               const float* __restrict__ pe,
                               __bf16* __restrict__ kh,
                               __bf16* __restrict__ vt) {
  int tid = blockIdx.x * 256 + threadIdx.x;
  int d2 = tid & 63;
  int h  = (tid >> 6) & 15;
  int s  = (tid >> 10) & 2047;
  int b  = (tid >> 21) & 1;
  size_t krow = ((size_t)(b * SS + s)) * (2 * DOUT);
  float k1 = bf2f(kvp[krow + h * HD + d2]);
  float k2 = bf2f(kvp[krow + h * HD + d2 + 64]);
  float v1 = bf2f(kvp[krow + DOUT + h * HD + d2]);
  float v2 = bf2f(kvp[krow + DOUT + h * HD + d2 + 64]);
  float p1 = pe[s * HD + d2], p2 = pe[s * HD + d2 + 64];
  size_t kout = (((size_t)(b * HEADS + h)) * SS + s) * HD + d2;
  kh[kout]      = f2bf(k1 * __cosf(p1) - k2 * __sinf(p1));
  kh[kout + 64] = f2bf(k2 * __cosf(p2) + k1 * __sinf(p2));
  size_t vout = (((size_t)(b * HEADS + h)) * HD + d2) * SS + s;
  vt[vout]                   = f2bf(v1);
  vt[vout + (size_t)64 * SS] = f2bf(v2);
}

// ---------------------------------------------------------------------------
// Flash attention, causal, scale = 1/sqrt(S).
// grid = (S/128, B*H). Block = 256 threads = 8 waves; wave w owns q rows
// [q0 + 16w, q0 + 16w + 16). Key tiles of 64 double-buffered in LDS via
// async copies; P round-trips through LDS to convert C-layout -> A-layout.
// ---------------------------------------------------------------------------
#define LDK 144  // sK row stride (64 keys x 128 d)
#define LDV 80   // sV row stride (128 d x 64 keys), also sP stride (16 x 64)

__global__ __launch_bounds__(256)
void mla_flash_attn_kernel(const __bf16* __restrict__ qh,
                           const __bf16* __restrict__ khp,
                           const __bf16* __restrict__ vtp,
                           __bf16* __restrict__ attn_out) {
  __shared__ __bf16 sK[2][64 * LDK];
  __shared__ __bf16 sV[2][128 * LDV];
  __shared__ __bf16 sP[8 * 16 * LDV];

  const int bh = blockIdx.y;
  const int b = bh >> 4, h = bh & 15;
  const int q0 = blockIdx.x * 128;
  const __bf16* Q  = qh  + (size_t)bh * SS * HD;
  const __bf16* Kp = khp + (size_t)bh * SS * HD;
  const __bf16* Vt = vtp + (size_t)bh * HD * SS;

  const int tid = threadIdx.x;
  const int w = tid >> 5, lane = tid & 31, l = lane & 15, hi = lane >> 4;
  const int qrow = q0 + w * 16;
  const float scale = 0.022097086912079608f;  // 1/sqrt(2048)

  // Q fragments for this wave's 16 rows (head dim 128 = 4 K-steps of 32)
  bf16x16 qf[4];
#pragma unroll
  for (int kd = 0; kd < 4; ++kd)
    qf[kd] = fragA(Q + (size_t)(qrow + l) * HD + kd * 32, hi);

  f32x8 zero = {0.f, 0.f, 0.f, 0.f, 0.f, 0.f, 0.f, 0.f};
  f32x8 o[8];
#pragma unroll
  for (int ds = 0; ds < 8; ++ds) o[ds] = zero;
  float mr[8], lr[8];
#pragma unroll
  for (int r = 0; r < 8; ++r) { mr[r] = -3.0e38f; lr[r] = 0.f; }

  __bf16* pw = sP + w * (16 * LDV);
  const int kend = q0 + 128;  // causal upper bound for this q tile

  // K tile [64][128] and Vt tile [128][64]: 1024 chunks each, 4 per thread.
  auto stageKV = [&](int k0, int bf) {
#pragma unroll
    for (int i = 0; i < 4; ++i) {
      int c = tid + i * 256;
      { int row = c >> 4, cc = (c & 15) * 8;
        stage16(Kp + (size_t)(k0 + row) * HD + cc, &sK[bf][row * LDK + cc]); }
      { int row = c >> 3, cc = (c & 7) * 8;
        stage16(Vt + (size_t)row * SS + k0 + cc, &sV[bf][row * LDV + cc]); }
    }
  };

  stageKV(0, 0);
  stage_wait();
  __syncthreads();

  int buf = 0;
  for (int k0 = 0; k0 < kend; k0 += 64) {
    if (k0 + 64 < kend) stageKV(k0 + 64, buf ^ 1);  // streams in under WMMAs

    // S = Q K^T for 64 keys (4 N-subtiles x 4 head-dim K-steps)
    f32x8 sacc[4];
#pragma unroll
    for (int ni = 0; ni < 4; ++ni) sacc[ni] = zero;
#pragma unroll
    for (int ni = 0; ni < 4; ++ni) {
#pragma unroll
      for (int kd = 0; kd < 4; ++kd) {
        bf16x16 bf = fragB(&sK[buf][(ni * 16 + l) * LDK + kd * 32], hi);
        sacc[ni] = wmma_bf16(qf[kd], bf, sacc[ni]);
      }
    }
    // scale + causal mask
#pragma unroll
    for (int ni = 0; ni < 4; ++ni)
#pragma unroll
      for (int r = 0; r < 8; ++r) {
        int key = k0 + ni * 16 + l;
        int row = qrow + hi * 8 + r;
        float v = sacc[ni][r] * scale;
        sacc[ni][r] = (key > row) ? -3.0e38f : v;
      }
    // online softmax: row stats per (half-wave, vgpr r); shfl_xor over masks
    // 1,2,4,8 reduces across the 16 lanes of each half (wave32).
#pragma unroll
    for (int r = 0; r < 8; ++r) {
      float rm = fmaxf(fmaxf(sacc[0][r], sacc[1][r]),
                       fmaxf(sacc[2][r], sacc[3][r]));
#pragma unroll
      for (int xm = 1; xm <= 8; xm <<= 1)
        rm = fmaxf(rm, __shfl_xor(rm, xm, 32));
      float mnew = fmaxf(mr[r], rm);
      float corr = __expf(mr[r] - mnew);
      mr[r] = mnew;
      float rs = 0.f;
#pragma unroll
      for (int ni = 0; ni < 4; ++ni) {
        float e = __expf(sacc[ni][r] - mnew);
        sacc[ni][r] = e;
        rs += e;
      }
#pragma unroll
      for (int xm = 1; xm <= 8; xm <<= 1) rs += __shfl_xor(rs, xm, 32);
      lr[r] = lr[r] * corr + rs;
#pragma unroll
      for (int ds = 0; ds < 8; ++ds) o[ds][r] *= corr;
    }
    // P (C-layout) -> LDS -> A-layout fragments (wave-private region)
#pragma unroll
    for (int ni = 0; ni < 4; ++ni)
#pragma unroll
      for (int r = 0; r < 8; ++r)
        pw[(hi * 8 + r) * LDV + ni * 16 + l] = f2bf(sacc[ni][r]);

    bf16x16 pf[2];
    pf[0] = fragA(pw + l * LDV, hi);
    pf[1] = fragA(pw + l * LDV + 32, hi);
    // O += P V  (8 d-subtiles x 2 key K-steps)
#pragma unroll
    for (int ds = 0; ds < 8; ++ds) {
      bf16x16 b0 = fragB(&sV[buf][(ds * 16 + l) * LDV], hi);
      bf16x16 b1 = fragB(&sV[buf][(ds * 16 + l) * LDV + 32], hi);
      o[ds] = wmma_bf16(pf[0], b0, o[ds]);
      o[ds] = wmma_bf16(pf[1], b1, o[ds]);
    }

    stage_wait();
    __syncthreads();
    buf ^= 1;
  }

  // normalize and write out in [B][S][DOUT] layout (A-matrix of final GEMM)
  size_t obase = ((size_t)b * SS) * DOUT + (size_t)h * HD;
#pragma unroll
  for (int ds = 0; ds < 8; ++ds)
#pragma unroll
    for (int r = 0; r < 8; ++r) {
      int row = qrow + hi * 8 + r;
      int d = ds * 16 + l;
      attn_out[obase + (size_t)row * DOUT + d] = f2bf(o[ds][r] / lr[r]);
    }
}

// ---------------------------------------------------------------------------
// Host launcher
// ---------------------------------------------------------------------------
extern "C" void kernel_launch(void* const* d_in, const int* in_sizes, int n_in,
                              void* d_out, int out_size, void* d_ws,
                              size_t ws_size, hipStream_t stream) {
  (void)in_sizes; (void)n_in; (void)out_size; (void)ws_size;
  const float* x  = (const float*)d_in[0];
  const float* pe = (const float*)d_in[1];
  const float* Wq = (const float*)d_in[2];
  const float* Wl = (const float*)d_in[3];
  const float* Wu = (const float*)d_in[4];
  const float* Wp = (const float*)d_in[5];
  float* out = (float*)d_out;

  char* ws = (char*)d_ws;
  size_t off = 0;
  auto take = [&](size_t elems) {
    __bf16* p = (__bf16*)(ws + off);
    off += ((elems * 2 + 255) & ~(size_t)255);
    return p;
  };
  const size_t M = (size_t)BB * SS;  // 4096
  __bf16* xb  = take(M * DIN);
  __bf16* wqT = take((size_t)DOUT * DIN);
  __bf16* wlT = take((size_t)LAT * DIN);
  __bf16* wuT = take((size_t)2 * DOUT * LAT);
  __bf16* wpT = take((size_t)DOUT * DOUT);
  __bf16* qp  = take(M * DOUT);
  __bf16* kvl = take(M * LAT);
  __bf16* kvp = take(M * 2 * DOUT);
  __bf16* qh  = take(M * DOUT);
  __bf16* kh  = take(M * DOUT);
  __bf16* vth = take(M * DOUT);
  __bf16* ao  = take(M * DOUT);

  dim3 blk(256);

  cvt_bf16_kernel<<<(int)(M * DIN / 4 / 256), blk, 0, stream>>>(x, xb,
                                                   (int)(M * DIN / 4));
  transpose_bf16_kernel<<<dim3(DOUT / 32, DIN / 32), blk, 0, stream>>>(
      Wq, wqT, DIN, DOUT);
  transpose_bf16_kernel<<<dim3(LAT / 32, DIN / 32), blk, 0, stream>>>(
      Wl, wlT, DIN, LAT);
  transpose_bf16_kernel<<<dim3(2 * DOUT / 32, LAT / 32), blk, 0, stream>>>(
      Wu, wuT, LAT, 2 * DOUT);
  transpose_bf16_kernel<<<dim3(DOUT / 32, DOUT / 32), blk, 0, stream>>>(
      Wp, wpT, DOUT, DOUT);

  // q = x @ Wq  (bf16 out, RoPE reads it)
  gemm_bf16_kernel<1><<<dim3(DOUT / 128, M / 128), blk, 0, stream>>>(
      xb, wqT, qp, (int)M, DOUT, DIN);
  // kv_latent = x @ Wl
  gemm_bf16_kernel<1><<<dim3(LAT / 128, M / 128), blk, 0, stream>>>(
      xb, wlT, kvl, (int)M, LAT, DIN);
  // kv = kv_latent @ Wu
  gemm_bf16_kernel<1><<<dim3(2 * DOUT / 128, M / 128), blk, 0, stream>>>(
      kvl, wuT, kvp, (int)M, 2 * DOUT, LAT);

  rope_q_kernel<<<16384, blk, 0, stream>>>(qp, pe, qh);
  rope_kv_kernel<<<16384, blk, 0, stream>>>(kvp, pe, kh, vth);

  mla_flash_attn_kernel<<<dim3(SS / 128, BB * HEADS), blk, 0, stream>>>(
      qh, kh, vth, ao);

  // out = attn @ Wp  (fp32 out)
  gemm_bf16_kernel<0><<<dim3(DOUT / 128, M / 128), blk, 0, stream>>>(
      ao, wpT, out, (int)M, DOUT, DOUT);
}
